// PopNet_807453852011
// MI455X (gfx1250) — matrix operand order repaired and checked by reading, not compiled, not run on hardware
//
#include <hip/hip_runtime.h>
#include <hip/hip_bf16.h>
#include <math.h>

typedef __attribute__((ext_vector_type(16))) _Float16 v16h;
typedef __attribute__((ext_vector_type(8)))  float    v8f;

namespace {

constexpr int N_  = 5000;
constexpr int T_  = 8;
constexpr int NT_ = N_ * T_;

__device__ __forceinline__ float dsig(float x) { return 1.f / (1.f + __expf(-x)); }
__device__ __forceinline__ float delu(float x) { return x > 0.f ? x : __expf(x) - 1.f; }

//================ WMMA GEMM: C = act(concat(A0,A1,A2) @ W^T + b) ==============
// Fully specialized on K-segment sizes, Kout and activation (all compile-time
// at every call site). The chunk loop unrolls; segment selection, K-tail
// tests, activation select and store strides all constant-fold, leaving
// straight-line aligned float4 loads (global_load_b128) + packed f32->f16
// converts + v_wmma_f32_16x16x32_f16 + branchless epilogue. Row index is
// clamped instead of predicated (A row r only feeds D row r; D rows >= M are
// never stored); full row-tiles take an unguarded store fast path.
template <int K0, int K1, int K2, int KOUT, int ACT>
__global__ __launch_bounds__(128)
void k_gemm(const float* __restrict__ A0, int ld0,
            const float* __restrict__ A1, int ld1,
            const float* __restrict__ A2, int ld2,
            const float* __restrict__ W, const float* __restrict__ bias,
            float* __restrict__ C, int M)
{
  constexpr int Ktot = K0 + K1 + K2;
  constexpr int K01  = K0 + K1;
  constexpr int NCH  = (Ktot + 31) / 32;
  const int lane    = threadIdx.x & 31;
  const int coltile = blockIdx.y * 4 + (threadIdx.x >> 5);
  if (coltile * 16 >= KOUT) return;                 // uniform per wave
  const int rowbase = blockIdx.x << 4;
  int row           = rowbase + (lane & 15);
  row               = (row < M) ? row : (M - 1);    // clamp, not predicate
  const int col     = (coltile << 4) + (lane & 15);
  const int hi      = lane >> 4;
  const int hi8     = hi << 3;
  const float* __restrict__ aw0 = A0 + (size_t)row * ld0;
  const float* __restrict__ aw1 = (K1 > 0) ? (A1 + (size_t)row * ld1) : A0;
  const float* __restrict__ aw2 = (K2 > 0) ? (A2 + (size_t)row * ld2) : A0;
  const float* __restrict__ brow = W + (size_t)col * Ktot;
  v8f acc = {};
#pragma unroll
  for (int ch = 0; ch < NCH; ++ch) {
    const int kc = ch * 32;
    v16h af, bf;
    // ---- A fragment: two 8-element runs; segment + bounds compile-time ----
#pragma unroll
    for (int rrun = 0; rrun < 2; ++rrun) {
      const int base = kc + (rrun << 4);            // compile-time constant
      const int eb = rrun << 3;
      if (base < Ktot) {
        const float* p;
        if (base < K0)       p = aw0 + base + hi8;
        else if (base < K01) p = aw1 + (base - K0) + hi8;
        else                 p = aw2 + (base - K01) + hi8;
        float4 lo  = ((const float4*)p)[0];
        float4 h4  = ((const float4*)p)[1];
        af[eb + 0] = (_Float16)lo.x; af[eb + 1] = (_Float16)lo.y;
        af[eb + 2] = (_Float16)lo.z; af[eb + 3] = (_Float16)lo.w;
        af[eb + 4] = (_Float16)h4.x; af[eb + 5] = (_Float16)h4.y;
        af[eb + 6] = (_Float16)h4.z; af[eb + 7] = (_Float16)h4.w;
      } else {
#pragma unroll
        for (int e = 0; e < 8; ++e) af[eb + e] = (_Float16)0.f;
      }
    }
    // ---- B fragment: two 8-element runs per lane (k = 8h+s, 16+8h+s) ----
    if (kc + 16 < Ktot) {                           // both hi halves valid
      const float4* p = (const float4*)(brow + kc + (hi << 3));
      float4 b0 = p[0], b1 = p[1], b2 = p[4], b3 = p[5];
      bf[0]  = (_Float16)b0.x; bf[1]  = (_Float16)b0.y;
      bf[2]  = (_Float16)b0.z; bf[3]  = (_Float16)b0.w;
      bf[4]  = (_Float16)b1.x; bf[5]  = (_Float16)b1.y;
      bf[6]  = (_Float16)b1.z; bf[7]  = (_Float16)b1.w;
      bf[8]  = (_Float16)b2.x; bf[9]  = (_Float16)b2.y;
      bf[10] = (_Float16)b2.z; bf[11] = (_Float16)b2.w;
      bf[12] = (_Float16)b3.x; bf[13] = (_Float16)b3.y;
      bf[14] = (_Float16)b3.z; bf[15] = (_Float16)b3.w;
    } else {                                        // 16-wide K tail: second run is OOB
      const float4* p = (const float4*)(brow + kc + (hi << 3));
      float4 b0 = p[0], b1 = p[1];
      bf[0]  = (_Float16)b0.x; bf[1]  = (_Float16)b0.y;
      bf[2]  = (_Float16)b0.z; bf[3]  = (_Float16)b0.w;
      bf[4]  = (_Float16)b1.x; bf[5]  = (_Float16)b1.y;
      bf[6]  = (_Float16)b1.z; bf[7]  = (_Float16)b1.w;
#pragma unroll
      for (int e = 8; e < 16; ++e) bf[e] = (_Float16)0.f;
    }
    acc = __builtin_amdgcn_wmma_f32_16x16x32_f16(false, af, false, bf,
                                                 (short)0, acc, false, false);
  }
  const float bb = bias[col];
  float* __restrict__ cp = C + (size_t)(rowbase + hi8) * KOUT + col;
  if (rowbase + 16 <= M) {                          // full tile: unguarded stores
#pragma unroll
    for (int r = 0; r < 8; ++r) {
      float v = acc[r] + bb;
      if (ACT == 1)      v = fmaxf(v, 0.f);
      else if (ACT == 2) v = delu(v);
      cp[(size_t)r * KOUT] = v;
    }
  } else {
#pragma unroll
    for (int r = 0; r < 8; ++r) {
      if (rowbase + r + hi8 < M) {
        float v = acc[r] + bb;
        if (ACT == 1)      v = fmaxf(v, 0.f);
        else if (ACT == 2) v = delu(v);
        cp[(size_t)r * KOUT] = v;
      }
    }
  }
}

//================ small elementwise / graph kernels ===========================
__global__ void k_fill(float* p, float v, int n) {
  int i = blockIdx.x * blockDim.x + threadIdx.x;
  if (i < n) p[i] = v;
}

// out[i*os] = dot(Z[i, :K], w)
__global__ void k_rowdot(const float* __restrict__ Z, int ldz,
                         const float* __restrict__ w, int K,
                         float* __restrict__ out, int os, int n) {
  int i = blockIdx.x * blockDim.x + threadIdx.x;
  if (i >= n) return;
  const float* z = Z + (size_t)i * ldz;
  float s = 0.f;
  for (int c = 0; c < K; ++c) s += z[c] * w[c];
  out[(size_t)i * os] = s;
}

// e = leaky(s1[src]+s2[dst]+b); segment-max into m[dst] (float via int tricks)
__global__ void k_edge_score(const int* __restrict__ src, const int* __restrict__ dst,
                             const float* __restrict__ s1, const float* __restrict__ s2,
                             const float* __restrict__ b, float* __restrict__ e,
                             float* __restrict__ m, int E) {
  int k = blockIdx.x * blockDim.x + threadIdx.x;
  if (k >= E) return;
  float v = s1[src[k]] + s2[dst[k]] + b[0];
  v = v >= 0.f ? v : 0.01f * v;
  e[k] = v;
  int d = dst[k];
  if (v >= 0.f) atomicMax((int*)&m[d], __float_as_int(v));
  else          atomicMin((unsigned int*)&m[d], (unsigned int)__float_as_int(v));
}

__global__ void k_edge_exp(const int* __restrict__ dst, float* __restrict__ e,
                           const float* __restrict__ m, float* __restrict__ ssum, int E) {
  int k = blockIdx.x * blockDim.x + threadIdx.x;
  if (k >= E) return;
  int d = dst[k];
  float ex = __expf(e[k] - m[d]);
  e[k] = ex;
  atomicAdd(&ssum[d], ex);
}

// out[dst] += (e/ssum[dst]) * val[src]  (block = 64 channels, grid = E edges)
__global__ void k_edge_agg(const int* __restrict__ src, const int* __restrict__ dst,
                           const float* __restrict__ e, const float* __restrict__ ssum,
                           const float* __restrict__ val, float* __restrict__ out, int E) {
  int k = blockIdx.x, c = threadIdx.x;
  if (k >= E) return;
  float a = e[k] / ssum[dst[k]];
  atomicAdd(&out[(size_t)dst[k] * 64 + c], a * val[(size_t)src[k] * 64 + c]);
}

__global__ void k_elu(const float* __restrict__ in, float* __restrict__ out, int n) {
  int i = blockIdx.x * blockDim.x + threadIdx.x;
  if (i < n) out[i] = delu(in[i]);
}

// u_node = elu(concat([u_node0, x_u_i]))  -> (N,128)
__global__ void k_unode(const float* __restrict__ un0, const float* __restrict__ xu,
                        int ldx, float* __restrict__ out, int n) {
  int i = blockIdx.x * blockDim.x + threadIdx.x;
  int nn = i >> 7, c = i & 127;
  if (nn >= n) return;
  float v = (c < 64) ? un0[(size_t)nn * 64 + c] : xu[(size_t)nn * ldx + (c - 64)];
  out[i] = delu(v);
}

// GRU combine; writes new h in place and into hist (stride T*64)
__global__ void k_gru(const float* __restrict__ gi, const float* __restrict__ gh,
                      float* __restrict__ h, float* __restrict__ hist, int n) {
  int i = blockIdx.x * blockDim.x + threadIdx.x;
  int nn = i >> 6, c = i & 63;
  if (nn >= n) return;
  const float* gir = gi + (size_t)nn * 192;
  const float* ghr = gh + (size_t)nn * 192;
  float r = dsig(gir[c] + ghr[c]);
  float z = dsig(gir[64 + c] + ghr[64 + c]);
  float a = tanhf(gir[128 + c] + r * ghr[128 + c]);
  float hn = (1.f - z) * a + z * h[i];
  h[i] = hn;
  hist[(size_t)nn * (T_ * 64) + c] = hn;
}

// causal dilated conv bank -> (N,T,48), dilations {1,3,5}, kw=3
__global__ void k_conv(const float* __restrict__ x,
                       const float* __restrict__ w1, const float* __restrict__ b1,
                       const float* __restrict__ w2, const float* __restrict__ b2,
                       const float* __restrict__ w3, const float* __restrict__ b3,
                       float* __restrict__ out, int n) {
  int idx = blockIdx.x * blockDim.x + threadIdx.x;
  int c = idx % 48, t = (idx / 48) % T_, nn = idx / (48 * T_);
  if (nn >= n) return;
  int s = c >> 4, oc = c & 15;
  int dil = (s == 0) ? 1 : (s == 1) ? 3 : 5;
  const float* w = (s == 0) ? w1 : (s == 1) ? w2 : w3;
  const float* b = (s == 0) ? b1 : (s == 1) ? b2 : b3;
  float acc = b[oc];
  const float* xr = x + (size_t)nn * T_ * 64;
  for (int k = 0; k < 3; ++k) {
    int tt = t + (k - 2) * dil;
    if (tt < 0) continue;
    const float* xc = xr + (size_t)tt * 64;
    const float* wc = w + (size_t)oc * 64 * 3 + k;
    for (int ci = 0; ci < 64; ++ci) acc += xc[ci] * wc[ci * 3];
  }
  out[idx] = acc;
}

// out[n,t,c] = mean over t'<=t of x[n,t',c]  (x: (N,T,64))
__global__ void k_cummean(const float* __restrict__ x, float* __restrict__ out, int n) {
  int i = blockIdx.x * blockDim.x + threadIdx.x;
  int nn = i >> 6, c = i & 63;
  if (nn >= n) return;
  float s = 0.f;
  for (int t = 0; t < T_; ++t) {
    size_t o = (size_t)nn * T_ * 64 + (size_t)t * 64 + c;
    s += x[o];
    out[o] = s / (float)(t + 1);
  }
}

// tie[n,t,c] = sigmoid(s48[n,t,c]) * mean_{t'<=t} conv[n,t',c]
__global__ void k_tie(const float* __restrict__ conv, const float* __restrict__ s48,
                      float* __restrict__ tie, int n) {
  int i = blockIdx.x * blockDim.x + threadIdx.x;
  int nn = i / 48, c = i % 48;
  if (nn >= n) return;
  float s = 0.f;
  for (int t = 0; t < T_; ++t) {
    size_t o = (size_t)nn * T_ * 48 + (size_t)t * 48 + c;
    s += conv[o];
    tie[o] = dsig(s48[o]) * (s / (float)(t + 1));
  }
}

// v[c] = sum_o utatt_W[o, c] * ta_W[o]   (fused attention projection)
__global__ void k_makev(const float* __restrict__ utattW, const float* __restrict__ taW,
                        float* __restrict__ v) {
  int c = threadIdx.x;
  if (c >= 64) return;
  float s = 0.f;
  for (int o = 0; o < 64; ++o) s += utattW[(size_t)o * 112 + c] * taW[o];
  v[c] = s;
}

// temporal attention over j<=step: hu_agg[n,step,:] = sum_j softmax(us[n,j]) * his[n,j,:]
__global__ void k_tattn(const float* __restrict__ us, const float* __restrict__ his,
                        float* __restrict__ hu_agg, int step, int n) {
  int i = blockIdx.x * blockDim.x + threadIdx.x;
  int nn = i >> 6, c = i & 63;
  if (nn >= n) return;
  const float* u = us + (size_t)nn * T_;
  float mx = -__builtin_inff();
  for (int j = 0; j <= step; ++j) mx = fmaxf(mx, u[j]);
  float sum = 0.f, acc = 0.f;
  for (int j = 0; j <= step; ++j) {
    float w = __expf(u[j] - mx);
    sum += w;
    acc += w * his[(size_t)nn * T_ * 64 + (size_t)j * 64 + c];
  }
  hu_agg[(size_t)nn * T_ * 64 + (size_t)step * 64 + c] = acc / sum;
}

__global__ void k_outdot(const float* __restrict__ hid, const float* __restrict__ w,
                         const float* __restrict__ b, float* __restrict__ out, int rows) {
  int r = blockIdx.x * blockDim.x + threadIdx.x;
  if (r >= rows) return;
  const float* h = hid + (size_t)r * 64;
  float s = b[0];
  for (int c = 0; c < 64; ++c) s += h[c] * w[c];
  out[r] = s;
}

// per-row KL contribution: sum_c softmax(raw)_c * (log_softmax(raw)_c - log_softmax(est)_c)
__global__ void k_kl(const float* __restrict__ raw, const float* __restrict__ est,
                     float* __restrict__ out, int rows, float scale) {
  int r = blockIdx.x * blockDim.x + threadIdx.x;
  if (r >= rows) return;
  const float* a = raw + (size_t)r * 64;
  const float* e = est + (size_t)r * 64;
  float ma = -__builtin_inff(), me = -__builtin_inff();
  for (int c = 0; c < 64; ++c) { ma = fmaxf(ma, a[c]); me = fmaxf(me, e[c]); }
  float sa = 0.f, se = 0.f;
  for (int c = 0; c < 64; ++c) { sa += __expf(a[c] - ma); se += __expf(e[c] - me); }
  float la = logf(sa), le = logf(se);
  float acc = 0.f;
  for (int c = 0; c < 64; ++c) {
    float p = __expf(a[c] - ma) / sa;
    acc += p * ((a[c] - ma - la) - (e[c] - me - le));
  }
  atomicAdd(out, acc * scale);
}

} // namespace

// host-side dispatch: one template instantiation per GEMM shape in the model
template <int K0, int K1, int K2, int KOUT, int ACT>
static void gemmT(hipStream_t s,
                  const float* A0, int ld0, const float* A1, int ld1,
                  const float* A2, int ld2, const float* W, const float* b,
                  float* C, int M) {
  dim3 g((M + 15) / 16, (KOUT + 63) / 64);
  k_gemm<K0, K1, K2, KOUT, ACT><<<g, 128, 0, s>>>(A0, ld0, A1, ld1, A2, ld2, W, b, C, M);
}

extern "C" void kernel_launch(void* const* d_in, const int* in_sizes, int n_in,
                              void* d_out, int out_size, void* d_ws, size_t ws_size,
                              hipStream_t stream) {
  const float* x_r  = (const float*)d_in[0];
  const float* x_u  = (const float*)d_in[1];
  const float* x_st = (const float*)d_in[2];
  const int*   src  = (const int*)d_in[5];
  const int*   dst  = (const int*)d_in[6];
  const int E = in_sizes[5];
  const float* P[80];
  for (int i = 7; i < n_in && i < 80; ++i) P[i] = (const float*)d_in[i];
  // param indices
  const float *gr_W=P[7],*gr_b=P[8],*gr_aW=P[9],*gr_ab=P[10];
  const float *gu_W=P[11],*gu_b=P[12],*gu_aW=P[13],*gu_ab=P[14];
  const float *cs_W=P[15],*cs_b=P[16],*cr_W=P[17],*cr_b=P[18];
  const float *cu_W=P[19],*cu_b=P[20],*ca_W=P[21],*ca_b=P[22];
  const float *grur_Wih=P[23],*grur_Whh=P[24],*grur_bih=P[25],*grur_bhh=P[26];
  const float *gruu_Wih=P[27],*gruu_Whh=P[28],*gruu_bih=P[29],*gruu_bhh=P[30];
  const float *cr1w=P[31],*cr1b=P[32],*cr2w=P[33],*cr2b=P[34],*cr3w=P[35],*cr3b=P[36];
  const float *cu1w=P[37],*cu1b=P[38],*cu2w=P[39],*cu2b=P[40],*cu3w=P[41],*cu3b=P[42];
  const float *rt1_W=P[43],*rt1_b=P[44],*rt2_W=P[45],*rt2_b=P[46];
  const float *ut1_W=P[47],*ut1_b=P[48],*ut2_W=P[49],*ut2_b=P[50];
  const float *utatt_W=P[53],*ta_W=P[55];
  const float *rfc_W=P[57],*rfc_b=P[58],*rout_W=P[59],*rout_b=P[60];
  const float *ufc_W=P[61],*ufc_b=P[62],*uout_W=P[63],*uout_b=P[64];
  const float *klr1_W=P[65],*klr1_b=P[66],*klr2_W=P[67],*klr2_b=P[68];
  const float *klu1_W=P[69],*klu1_b=P[70],*klu2_W=P[71],*klu2_b=P[72];

  // ---- workspace carve-out (floats) ----
  float* w = (float*)d_ws;
  auto take = [&](size_t n) { float* p = w; w += n; return p; };
  float* sie     = take((size_t)N_ * 32);
  float* convT_r = take((size_t)NT_ * 48);
  float* convT_u = take((size_t)NT_ * 48);
  float* r_tie   = take((size_t)NT_ * 48);
  float* u_tie   = take((size_t)NT_ * 48);
  float* his_hr  = take((size_t)NT_ * 64);
  float* his_hu  = take((size_t)NT_ * 64);
  float* hu_agg  = take((size_t)NT_ * 64);
  float* h_r     = take((size_t)N_ * 64);
  float* h_u     = take((size_t)N_ * 64);
  float* zbuf    = take((size_t)N_ * 64);
  float* r_node  = take((size_t)N_ * 64);
  float* u_node0 = take((size_t)N_ * 64);
  float* agg     = take((size_t)N_ * 64);
  float* agg_u   = take((size_t)N_ * 64);
  float* z1c     = take((size_t)N_ * 64);
  float* z2c     = take((size_t)N_ * 64);
  float* u_node  = take((size_t)N_ * 128);
  float* gi      = take((size_t)N_ * 192);
  float* gh      = take((size_t)N_ * 192);
  float* s1      = take(N_);
  float* s2      = take(N_);
  float* mseg    = take(N_);
  float* ssum    = take(N_);
  float* e_edge  = take(E);
  float* u_s     = take(NT_);
  float* v64     = take(64);
  float* meanx   = take((size_t)NT_ * 64);
  float* hid16   = take((size_t)NT_ * 16);
  float* s48     = take((size_t)NT_ * 48);
  float* big_hid = take((size_t)NT_ * 64);
  float* est     = take((size_t)NT_ * 64);
  (void)ws_size;

  float* outp = (float*)d_out; (void)out_size;
  const float* nul = nullptr;
  auto elw = [&](int n) { return dim3((n + 255) / 256); };

  // ---- state init + precompute ----
  hipMemcpyAsync(h_r, d_in[3], (size_t)N_ * 64 * sizeof(float), hipMemcpyDeviceToDevice, stream);
  hipMemcpyAsync(h_u, d_in[4], (size_t)N_ * 64 * sizeof(float), hipMemcpyDeviceToDevice, stream);
  gemmT<32,0,0,32,0>(stream, x_st, 32, nul, 0, nul, 0, cs_W, cs_b, sie, N_);
  k_conv<<<elw(NT_ * 48), 256, 0, stream>>>(x_r, cr1w, cr1b, cr2w, cr2b, cr3w, cr3b, convT_r, N_);
  k_conv<<<elw(NT_ * 48), 256, 0, stream>>>(x_u, cu1w, cu1b, cu2w, cu2b, cu3w, cu3b, convT_u, N_);
  k_makev<<<1, 64, 0, stream>>>(utatt_W, ta_W, v64);
  // tie paths (r then u, reusing meanx/hid16/s48)
  k_cummean<<<elw(N_ * 64), 256, 0, stream>>>(x_r, meanx, N_);
  gemmT<64,0,0,16,1>(stream, meanx, 64, nul, 0, nul, 0, rt1_W, rt1_b, hid16, NT_);
  gemmT<16,0,0,48,0>(stream, hid16, 16, nul, 0, nul, 0, rt2_W, rt2_b, s48, NT_);
  k_tie<<<elw(N_ * 48), 256, 0, stream>>>(convT_r, s48, r_tie, N_);
  k_cummean<<<elw(N_ * 64), 256, 0, stream>>>(x_u, meanx, N_);
  gemmT<64,0,0,16,1>(stream, meanx, 64, nul, 0, nul, 0, ut1_W, ut1_b, hid16, NT_);
  gemmT<16,0,0,48,0>(stream, hid16, 16, nul, 0, nul, 0, ut2_W, ut2_b, s48, NT_);
  k_tie<<<elw(N_ * 48), 256, 0, stream>>>(convT_u, s48, u_tie, N_);

  // ---- time loop ----
  for (int i = 0; i < T_; ++i) {
    const float* xr_i = x_r + (size_t)i * 64;   // (N,T,64) slice, row stride T*64
    const float* xu_i = x_u + (size_t)i * 64;
    // GAT r
    gemmT<64,0,0,64,0>(stream, xr_i, T_ * 64, nul, 0, nul, 0, gr_W, gr_b, zbuf, N_);
    k_rowdot<<<elw(N_), 256, 0, stream>>>(zbuf, 64, gr_aW, 64, s1, 1, N_);
    k_rowdot<<<elw(N_), 256, 0, stream>>>(zbuf, 64, gr_aW + 64, 64, s2, 1, N_);
    k_fill<<<elw(N_), 256, 0, stream>>>(mseg, -__builtin_inff(), N_);
    k_fill<<<elw(N_), 256, 0, stream>>>(ssum, 0.f, N_);
    k_fill<<<elw(N_ * 64), 256, 0, stream>>>(agg, 0.f, N_ * 64);
    k_edge_score<<<elw(E), 256, 0, stream>>>(src, dst, s1, s2, gr_ab, e_edge, mseg, E);
    k_edge_exp<<<elw(E), 256, 0, stream>>>(dst, e_edge, mseg, ssum, E);
    k_edge_agg<<<E, 64, 0, stream>>>(src, dst, e_edge, ssum, zbuf, agg, E);
    k_elu<<<elw(N_ * 64), 256, 0, stream>>>(agg, r_node, N_ * 64);
    // GAT u
    gemmT<64,0,0,64,0>(stream, xu_i, T_ * 64, nul, 0, nul, 0, gu_W, gu_b, zbuf, N_);
    k_rowdot<<<elw(N_), 256, 0, stream>>>(zbuf, 64, gu_aW, 64, s1, 1, N_);
    k_rowdot<<<elw(N_), 256, 0, stream>>>(zbuf, 64, gu_aW + 64, 64, s2, 1, N_);
    k_fill<<<elw(N_), 256, 0, stream>>>(mseg, -__builtin_inff(), N_);
    k_fill<<<elw(N_), 256, 0, stream>>>(ssum, 0.f, N_);
    k_fill<<<elw(N_ * 64), 256, 0, stream>>>(agg, 0.f, N_ * 64);
    k_edge_score<<<elw(E), 256, 0, stream>>>(src, dst, s1, s2, gu_ab, e_edge, mseg, E);
    k_edge_exp<<<elw(E), 256, 0, stream>>>(dst, e_edge, mseg, ssum, E);
    k_edge_agg<<<E, 64, 0, stream>>>(src, dst, e_edge, ssum, zbuf, agg, E);
    k_elu<<<elw(N_ * 64), 256, 0, stream>>>(agg, u_node0, N_ * 64);
    // cross attention: z1=[r_node,sie]@cr^T, z2=[u_node0,sie]@cu^T, score decomposed
    gemmT<64,32,0,64,0>(stream, r_node, 64, sie, 32, nul, 0, cr_W, cr_b, z1c, N_);
    gemmT<64,32,0,64,0>(stream, u_node0, 64, sie, 32, nul, 0, cu_W, cu_b, z2c, N_);
    k_rowdot<<<elw(N_), 256, 0, stream>>>(z2c, 64, ca_W, 64, s1, 1, N_); // src term
    k_rowdot<<<elw(N_), 256, 0, stream>>>(z1c, 64, ca_W, 64, s2, 1, N_); // dst term
    k_fill<<<elw(N_), 256, 0, stream>>>(mseg, -__builtin_inff(), N_);
    k_fill<<<elw(N_), 256, 0, stream>>>(ssum, 0.f, N_);
    k_fill<<<elw(N_ * 64), 256, 0, stream>>>(agg, 0.f, N_ * 64);
    k_edge_score<<<elw(E), 256, 0, stream>>>(src, dst, s1, s2, ca_b, e_edge, mseg, E);
    k_edge_exp<<<elw(E), 256, 0, stream>>>(dst, e_edge, mseg, ssum, E);
    k_edge_agg<<<E, 64, 0, stream>>>(src, dst, e_edge, ssum, u_node0, agg, E);
    k_elu<<<elw(N_ * 64), 256, 0, stream>>>(agg, agg_u, N_ * 64);
    // u_node = elu(concat([u_node0, xu_i]))
    k_unode<<<elw(N_ * 128), 256, 0, stream>>>(u_node0, xu_i, T_ * 64, u_node, N_);
    // GRU r  (input = concat([r_node, agg_u, xr_i]) via 3-segment GEMM)
    gemmT<64,64,64,192,0>(stream, r_node, 64, agg_u, 64, xr_i, T_ * 64, grur_Wih, grur_bih, gi, N_);
    gemmT<64,0,0,192,0>(stream, h_r, 64, nul, 0, nul, 0, grur_Whh, grur_bhh, gh, N_);
    k_gru<<<elw(N_ * 64), 256, 0, stream>>>(gi, gh, h_r, his_hr + (size_t)i * 64, N_);
    // GRU u
    gemmT<128,0,0,192,0>(stream, u_node, 128, nul, 0, nul, 0, gruu_Wih, gruu_bih, gi, N_);
    gemmT<64,0,0,192,0>(stream, h_u, 64, nul, 0, nul, 0, gruu_Whh, gruu_bhh, gh, N_);
    k_gru<<<elw(N_ * 64), 256, 0, stream>>>(gi, gh, h_u, his_hu + (size_t)i * 64, N_);
    // temporal attention: scores reduce to h_u_j . v (j-independent terms cancel in softmax)
    k_rowdot<<<elw(N_), 256, 0, stream>>>(h_u, 64, v64, 64, u_s + i, T_, N_);
    k_tattn<<<elw(N_ * 64), 256, 0, stream>>>(u_s, his_hu, hu_agg, i, N_);
  }

  // ---- KL term ----
  k_fill<<<1, 1, 0, stream>>>(outp + 2 * NT_, 0.f, 1);
  gemmT<48,0,0,64,1>(stream, convT_r, 48, nul, 0, nul, 0, klr1_W, klr1_b, big_hid, NT_);
  gemmT<64,0,0,64,0>(stream, big_hid, 64, nul, 0, nul, 0, klr2_W, klr2_b, est, NT_);
  k_kl<<<elw(NT_), 256, 0, stream>>>(his_hr, est, outp + 2 * NT_, NT_, 1.f / (float)NT_);
  gemmT<48,0,0,64,1>(stream, convT_u, 48, nul, 0, nul, 0, klu1_W, klu1_b, big_hid, NT_);
  gemmT<64,0,0,64,0>(stream, big_hid, 64, nul, 0, nul, 0, klu2_W, klu2_b, est, NT_);
  k_kl<<<elw(NT_), 256, 0, stream>>>(his_hu, est, outp + 2 * NT_, NT_, 1.f / (float)NT_);

  // ---- output heads ----
  gemmT<64,64,48,64,1>(stream, his_hr, 64, hu_agg, 64, r_tie, 48, rfc_W, rfc_b, big_hid, NT_);
  k_outdot<<<elw(NT_), 256, 0, stream>>>(big_hid, rout_W, rout_b, outp, NT_);
  gemmT<64,48,0,64,1>(stream, his_hu, 64, u_tie, 48, nul, 0, ufc_W, ufc_b, big_hid, NT_);
  k_outdot<<<elw(NT_), 256, 0, stream>>>(big_hid, uout_W, uout_b, outp + NT_, NT_);
}
